// DeepSeekV3_1MoE_2001454760762
// MI455X (gfx1250) — compile-verified, matrix operand order checked by reading, not run on hardware
//
#include <hip/hip_runtime.h>
#include <hip/hip_bf16.h>

// Problem constants (match reference)
#define H_DIM 1024
#define I_DIM 512
#define E_NUM 32
#define TOPK  8
#define T_TOK 4096   // B*S = 4*1024

// A-panel LDS row stride in halfs: 1024 data + 2 TDM pads of 8 halfs (16B)
// (pad_interval = 256 dwords, pad_amount = 4 dwords) -> 520 dwords/row.
#define A_STRIDE 1040

typedef __bf16 bf16_t;
typedef bf16_t v16bf __attribute__((ext_vector_type(16)));
typedef bf16_t v8bf  __attribute__((ext_vector_type(8)));
typedef bf16_t v4bf  __attribute__((ext_vector_type(4)));
typedef float  v8f   __attribute__((ext_vector_type(8)));
typedef float  v4f   __attribute__((ext_vector_type(4)));
typedef unsigned int v4u __attribute__((ext_vector_type(4)));
typedef int v8i __attribute__((ext_vector_type(8)));
typedef int v4i __attribute__((ext_vector_type(4)));

static __device__ __forceinline__ v16bf cat16(v8bf lo, v8bf hi) {
  return __builtin_shufflevector(lo, hi, 0,1,2,3,4,5,6,7,8,9,10,11,12,13,14,15);
}

static __device__ __forceinline__ v8f wmma_bf16(v16bf a, v16bf b, v8f c) {
  // (neg_a, A, neg_b, B, c_mod, C, reuse_a, reuse_b)
  return __builtin_amdgcn_wmma_f32_16x16x32_bf16(false, a, false, b, (short)0, c, false, false);
}

// ---------------------------------------------------------------------------
// Kernel 1: fp32 activations -> bf16
// ---------------------------------------------------------------------------
__global__ void cvt_x_kernel(const float* __restrict__ x, bf16_t* __restrict__ xbf, int n4) {
  int i = blockIdx.x * blockDim.x + threadIdx.x;
  if (i >= n4) return;
  v4f v = *(const v4f*)(x + (size_t)i * 4);
  v4bf o;
  o[0] = (bf16_t)v[0]; o[1] = (bf16_t)v[1]; o[2] = (bf16_t)v[2]; o[3] = (bf16_t)v[3];
  *(v4bf*)(xbf + (size_t)i * 4) = o;
}

// ---------------------------------------------------------------------------
// Kernel 2: fp32 [batch][R][C] row-major -> bf16 [batch][C][R] (n-major)
// ---------------------------------------------------------------------------
__global__ void transpose_cvt_kernel(const float* __restrict__ src, bf16_t* __restrict__ dst,
                                     int R, int C) {
  __shared__ float tile[32][33];
  const size_t bo = (size_t)blockIdx.z * R * C;
  src += bo; dst += bo;
  const int c0 = blockIdx.x * 32, r0 = blockIdx.y * 32;
  const int tx = threadIdx.x & 31, ty = threadIdx.x >> 5;   // 256 threads: ty 0..7
  #pragma unroll
  for (int i = 0; i < 32; i += 8)
    tile[ty + i][tx] = src[(size_t)(r0 + ty + i) * C + (c0 + tx)];
  __syncthreads();
  #pragma unroll
  for (int i = 0; i < 32; i += 8)
    dst[(size_t)(c0 + ty + i) * R + (r0 + tx)] = (bf16_t)tile[tx][ty + i];
}

// ---------------------------------------------------------------------------
// Kernel 3: router. One wave32 per token; lane == expert (E == 32).
// ---------------------------------------------------------------------------
__global__ void router_kernel(const float* __restrict__ x, const float* __restrict__ Wr,
                              const float* __restrict__ br, int* __restrict__ counts,
                              int* __restrict__ list, float* __restrict__ wlist) {
  const int lane = threadIdx.x & 31;
  const int wid  = threadIdx.x >> 5;
  const int t = blockIdx.x * 8 + wid;
  if (t >= T_TOK) return;
  const float* xr = x + (size_t)t * H_DIM;
  float acc = br[lane];
  #pragma unroll 8
  for (int h = 0; h < H_DIM; ++h)
    acc += xr[h] * Wr[(size_t)h * E_NUM + lane];   // lane-contiguous: coalesced
  const float aff = 1.0f / (1.0f + __expf(-acc));
  float cur = aff, sum = 0.0f;
  bool sel = false;
  #pragma unroll
  for (int it = 0; it < TOPK; ++it) {
    float m = cur; int mi = lane;
    #pragma unroll
    for (int off = 16; off > 0; off >>= 1) {
      float om = __shfl_xor(m, off);
      int   oi = __shfl_xor(mi, off);
      if (om > m || (om == m && oi < mi)) { m = om; mi = oi; }
    }
    sum += m;
    if (lane == mi) { sel = true; cur = -1e30f; }
  }
  if (sel) {
    const float w = aff / (sum + 1e-9f);
    const int pos = atomicAdd(&counts[lane], 1);
    list [(size_t)lane * T_TOK + pos] = t;
    wlist[(size_t)lane * T_TOK + pos] = w;
  }
}

// ---------------------------------------------------------------------------
// Kernel 4: fused SwiGLU expert FFN over a 32-token M-tile.
// A-panel (32 gathered token rows x H) is DMA'd into LDS by the Tensor Data
// Mover (gather mode, 16-bit row indices, 2 descriptors of 16 rows), with
// hardware LDS padding for bank-conflict mitigation. Stage 1 computes gate &
// up with v_wmma_f32_16x16x32_bf16 (B streamed straight from L2, n-major
// pre-transposed bf16), fuses silu(g)*u into an LDS bf16 tile; stage 2 runs
// the down projection and scatters w*(acc+bd) via global_atomic_add_f32.
// ---------------------------------------------------------------------------
__global__ __launch_bounds__(256, 1)
void expert_ffn_kernel(const bf16_t* __restrict__ xbf,
                       const bf16_t* __restrict__ WgT,   // [e][I][H]
                       const bf16_t* __restrict__ WuT,   // [e][I][H]
                       const bf16_t* __restrict__ WdT,   // [e][H][I]
                       const float*  __restrict__ bg,    // [e][I]
                       const float*  __restrict__ bu,    // [e][I]
                       const float*  __restrict__ bd,    // [e][H]
                       const int*    __restrict__ counts,
                       const int*    __restrict__ list,
                       const float*  __restrict__ wlist,
                       float* __restrict__ out,
                       int isShared) {
  __shared__ __align__(16) bf16_t As[32 * A_STRIDE];  // A panel, TDM-padded rows
  __shared__ __align__(16) bf16_t Act[32][520];       // silu(g)*u, padded rows
  __shared__ int   sTok[32];
  __shared__ float sW[32];

  const int e  = blockIdx.y;
  const int m0 = blockIdx.x * 32;
  const int cnt = isShared ? T_TOK : counts[e];
  if (m0 >= cnt) return;
  const int rows = (cnt - m0 < 32) ? (cnt - m0) : 32;

  const bf16_t* wg  = WgT + (size_t)e * I_DIM * H_DIM;
  const bf16_t* wu  = WuT + (size_t)e * I_DIM * H_DIM;
  const bf16_t* wd  = WdT + (size_t)e * H_DIM * I_DIM;
  const float*  bgp = bg + (size_t)e * I_DIM;
  const float*  bup = bu + (size_t)e * I_DIM;
  const float*  bdp = bd + (size_t)e * H_DIM;

  const int tid = threadIdx.x;
  if (tid < 32) {
    // per-lane token/weight for the epilogue
    int tok; float w;
    if (isShared)        { tok = m0 + tid; w = 1.0f; }
    else if (tid < rows) { tok = list[(size_t)e * T_TOK + m0 + tid];
                           w  = wlist[(size_t)e * T_TOK + m0 + tid]; }
    else                 { tok = list[(size_t)e * T_TOK + m0]; w = 0.0f; }
    sTok[tid] = tok; sW[tid] = w;

    // --- Tensor Data Mover: gather the 32 token rows into the LDS A panel ---
    const unsigned lds_base = (unsigned)(uintptr_t)&As[0];
    const unsigned long long ga = (unsigned long long)(uintptr_t)xbf;
    #pragma unroll
    for (int half = 0; half < 2; ++half) {
      v4u g0; v8i g1; v4i g2, g3;
      // group0: count=1, gather_mode=1 (16-bit indices), lds_addr, global_addr, type=2
      g0[0] = 1u | (1u << 31);
      g0[1] = lds_base + (unsigned)half * 16u * A_STRIDE * 2u;
      g0[2] = (unsigned)ga;
      g0[3] = (unsigned)(ga >> 32) | (2u << 30);
      // group1: data_size=2B, pad_enable, pad_interval=256 dwords, pad_amount=4 dwords
      g1[0] = (1 << 16) | (1 << 20) | (7 << 22) | (3 << 25);
      g1[1] = (int)((unsigned)(H_DIM & 0xffff) << 16);  // tensor_dim0[15:0]
      g1[2] = (int)((unsigned)(T_TOK & 0xffff) << 16);  // tensor_dim1[15:0]
      g1[3] = (int)((unsigned)H_DIM << 16);             // tile_dim0 = H
      g1[4] = 16;                                       // tile_dim1 = 16 valid indices
      g1[5] = H_DIM;                                    // tensor_dim0_stride
      g1[6] = 0; g1[7] = 0;
      // groups 2/3: 16 packed 16-bit row indices (gather mode)
      int idx[16];
      #pragma unroll
      for (int i2 = 0; i2 < 16; ++i2) {
        const int m = half * 16 + i2;
        int tok2;
        if (isShared) tok2 = m0 + m;
        else          tok2 = list[(size_t)e * T_TOK + m0 + ((m < rows) ? m : 0)];
        idx[i2] = tok2 & 0xffff;
      }
      #pragma unroll
      for (int j = 0; j < 4; ++j) {
        g2[j] = idx[2 * j]     | (idx[2 * j + 1] << 16);
        g3[j] = idx[8 + 2 * j] | (idx[9 + 2 * j] << 16);
      }
#if defined(__clang_major__) && (__clang_major__ >= 23)
      v8i gz = {0, 0, 0, 0, 0, 0, 0, 0};
      __builtin_amdgcn_tensor_load_to_lds(g0, g1, g2, g3, gz, 0);
#else
      __builtin_amdgcn_tensor_load_to_lds(g0, g1, g2, g3, 0);
#endif
    }
    __builtin_amdgcn_s_wait_tensorcnt(0);
  }
  __syncthreads();

  const int lane = tid & 31;
  const int wid  = tid >> 5;     // 0..7
  const int nlo  = lane & 15;
  const int hs   = lane >> 4;    // half-wave select

  const v8f vzero = {0.f,0.f,0.f,0.f,0.f,0.f,0.f,0.f};
  v8f accG[2][4], accU[2][4];
  #pragma unroll
  for (int mi = 0; mi < 2; ++mi)
    #pragma unroll
    for (int ni = 0; ni < 4; ++ni) { accG[mi][ni] = vzero; accU[mi][ni] = vzero; }

  // ---- Stage 1: gate & up GEMM over K = H (no barriers in the loop) ----
  for (int k0 = 0; k0 < H_DIM; k0 += 32) {
    const int kpad = k0 + ((k0 >> 9) << 3);   // skip TDM pad halfs
    v16bf a[2];
    #pragma unroll
    for (int mi = 0; mi < 2; ++mi) {
      const bf16_t* ap = &As[0] + (size_t)(mi * 16 + nlo) * A_STRIDE + kpad;
      a[mi] = cat16(*(const v8bf*)(ap + hs * 8), *(const v8bf*)(ap + 16 + hs * 8));
    }
    #pragma unroll
    for (int ni = 0; ni < 4; ++ni) {
      const int n = wid * 64 + ni * 16 + nlo;
      const bf16_t* pg = wg + (size_t)n * H_DIM + k0;
      const bf16_t* pu = wu + (size_t)n * H_DIM + k0;
      v16bf bG = cat16(*(const v8bf*)(pg + hs * 8), *(const v8bf*)(pg + 16 + hs * 8));
      v16bf bU = cat16(*(const v8bf*)(pu + hs * 8), *(const v8bf*)(pu + 16 + hs * 8));
      __builtin_prefetch(pg + 32, 0, 1);   // next K-chunk -> global_prefetch_b8
      __builtin_prefetch(pu + 32, 0, 1);
      #pragma unroll
      for (int mi = 0; mi < 2; ++mi) {
        accG[mi][ni] = wmma_bf16(a[mi], bG, accG[mi][ni]);
        accU[mi][ni] = wmma_bf16(a[mi], bU, accU[mi][ni]);
      }
    }
  }

  // ---- Epilogue 1: act = silu(g + bg) * (u + bu) -> LDS (bf16) ----
  #pragma unroll
  for (int mi = 0; mi < 2; ++mi)
    #pragma unroll
    for (int ni = 0; ni < 4; ++ni) {
      const int n = wid * 64 + ni * 16 + nlo;
      const float bgv = bgp[n], buv = bup[n];
      #pragma unroll
      for (int r = 0; r < 8; ++r) {
        const int m = mi * 16 + hs * 8 + r;
        const float g = accG[mi][ni][r] + bgv;
        const float u = accU[mi][ni][r] + buv;
        const float s = g / (1.0f + __expf(-g));
        Act[m][n] = (bf16_t)(s * u);
      }
    }
  __syncthreads();

  // ---- Stage 2: down GEMM over K = I ----
  v8f accD[2][8];
  #pragma unroll
  for (int mi = 0; mi < 2; ++mi)
    #pragma unroll
    for (int ni = 0; ni < 8; ++ni) accD[mi][ni] = vzero;

  for (int k0 = 0; k0 < I_DIM; k0 += 32) {
    v16bf a[2];
    #pragma unroll
    for (int mi = 0; mi < 2; ++mi) {
      const int r = mi * 16 + nlo;
      a[mi] = cat16(*(const v8bf*)&Act[r][k0 + hs * 8],
                    *(const v8bf*)&Act[r][k0 + 16 + hs * 8]);
    }
    #pragma unroll
    for (int ni = 0; ni < 8; ++ni) {
      const int n = wid * 128 + ni * 16 + nlo;
      const bf16_t* pd = wd + (size_t)n * I_DIM + k0;
      v16bf bD = cat16(*(const v8bf*)(pd + hs * 8), *(const v8bf*)(pd + 16 + hs * 8));
      __builtin_prefetch(pd + 32, 0, 1);
      #pragma unroll
      for (int mi = 0; mi < 2; ++mi)
        accD[mi][ni] = wmma_bf16(a[mi], bD, accD[mi][ni]);
    }
  }

  // ---- Epilogue 2: scale & scatter ----
  float bdv[8];
  #pragma unroll
  for (int ni = 0; ni < 8; ++ni) bdv[ni] = bdp[wid * 128 + ni * 16 + nlo];
  #pragma unroll
  for (int mi = 0; mi < 2; ++mi)
    #pragma unroll
    for (int r = 0; r < 8; ++r) {
      const int m = mi * 16 + hs * 8 + r;
      if (isShared) {
        const size_t ob = (size_t)(m0 + m) * H_DIM;
        #pragma unroll
        for (int ni = 0; ni < 8; ++ni) {
          const int n = wid * 128 + ni * 16 + nlo;
          out[ob + n] = accD[mi][ni][r] + bdv[ni];
        }
      } else if (m < rows) {
        const float w = sW[m];
        const size_t ob = (size_t)sTok[m] * H_DIM;
        #pragma unroll
        for (int ni = 0; ni < 8; ++ni) {
          const int n = wid * 128 + ni * 16 + nlo;
          atomicAdd(&out[ob + n], w * (accD[mi][ni][r] + bdv[ni]));
        }
      }
    }
}

// ---------------------------------------------------------------------------
// Host launcher
// ---------------------------------------------------------------------------
extern "C" void kernel_launch(void* const* d_in, const int* in_sizes, int n_in,
                              void* d_out, int out_size, void* d_ws, size_t ws_size,
                              hipStream_t stream) {
  const float* x    = (const float*)d_in[0];
  const float* Wr   = (const float*)d_in[1];
  const float* br   = (const float*)d_in[2];
  const float* Wg_s = (const float*)d_in[3];
  const float* bg_s = (const float*)d_in[4];
  const float* Wu_s = (const float*)d_in[5];
  const float* bu_s = (const float*)d_in[6];
  const float* Wd_s = (const float*)d_in[7];
  const float* bd_s = (const float*)d_in[8];
  const float* Wg   = (const float*)d_in[9];
  const float* bg   = (const float*)d_in[10];
  const float* Wu   = (const float*)d_in[11];
  const float* bu   = (const float*)d_in[12];
  const float* Wd   = (const float*)d_in[13];
  const float* bd   = (const float*)d_in[14];
  float* out = (float*)d_out;

  char* ws = (char*)d_ws;
  size_t off = 0;
  auto take = [&](size_t bytes) -> void* {
    void* p = ws + off;
    off += (bytes + 255) & ~(size_t)255;
    return p;
  };
  bf16_t* xbf    = (bf16_t*)take((size_t)T_TOK * H_DIM * 2);
  bf16_t* WgT    = (bf16_t*)take((size_t)E_NUM * I_DIM * H_DIM * 2);
  bf16_t* WuT    = (bf16_t*)take((size_t)E_NUM * I_DIM * H_DIM * 2);
  bf16_t* WdT    = (bf16_t*)take((size_t)E_NUM * H_DIM * I_DIM * 2);
  bf16_t* WgsT   = (bf16_t*)take((size_t)I_DIM * H_DIM * 2);
  bf16_t* WusT   = (bf16_t*)take((size_t)I_DIM * H_DIM * 2);
  bf16_t* WdsT   = (bf16_t*)take((size_t)H_DIM * I_DIM * 2);
  int*    counts = (int*)take(E_NUM * sizeof(int));
  int*    list   = (int*)take((size_t)E_NUM * T_TOK * sizeof(int));
  float*  wlist  = (float*)take((size_t)E_NUM * T_TOK * sizeof(float));

  // 1) activations -> bf16
  {
    const int n4 = T_TOK * H_DIM / 4;
    cvt_x_kernel<<<(n4 + 255) / 256, 256, 0, stream>>>(x, xbf, n4);
  }
  // 2) weight transpose + bf16 conversion (n-major), routed + shared
  transpose_cvt_kernel<<<dim3(I_DIM/32, H_DIM/32, E_NUM), 256, 0, stream>>>(Wg, WgT, H_DIM, I_DIM);
  transpose_cvt_kernel<<<dim3(I_DIM/32, H_DIM/32, E_NUM), 256, 0, stream>>>(Wu, WuT, H_DIM, I_DIM);
  transpose_cvt_kernel<<<dim3(H_DIM/32, I_DIM/32, E_NUM), 256, 0, stream>>>(Wd, WdT, I_DIM, H_DIM);
  transpose_cvt_kernel<<<dim3(I_DIM/32, H_DIM/32, 1),     256, 0, stream>>>(Wg_s, WgsT, H_DIM, I_DIM);
  transpose_cvt_kernel<<<dim3(I_DIM/32, H_DIM/32, 1),     256, 0, stream>>>(Wu_s, WusT, H_DIM, I_DIM);
  transpose_cvt_kernel<<<dim3(H_DIM/32, I_DIM/32, 1),     256, 0, stream>>>(Wd_s, WdsT, I_DIM, H_DIM);
  // 3) router (needs zeroed counts)
  hipMemsetAsync(counts, 0, E_NUM * sizeof(int), stream);
  router_kernel<<<T_TOK / 8, 256, 0, stream>>>(x, Wr, br, counts, list, wlist);
  // 4) shared expert: writes base values into out (full coverage)
  expert_ffn_kernel<<<dim3(T_TOK / 32, 1), 256, 0, stream>>>(
      xbf, WgsT, WusT, WdsT, bg_s, bu_s, bd_s, nullptr, nullptr, nullptr, out, 1);
  // 5) routed experts: atomic accumulate weighted contributions
  expert_ffn_kernel<<<dim3(T_TOK / 32, E_NUM), 256, 0, stream>>>(
      xbf, WgT, WuT, WdT, bg, bu, bd, counts, list, wlist, out, 0);
  (void)in_sizes; (void)n_in; (void)out_size; (void)ws_size;
}
